// ChebNet_66760971649326
// MI455X (gfx1250) — compile-verified, matrix operand order checked by reading, not run on hardware
//
#include <hip/hip_runtime.h>
#include <stdint.h>

// ---------------------------------------------------------------------------
// ChebNet on gfx1250: fp32 SpMM (atomics) + bf16 WMMA GEMMs (f32 accumulate)
// ---------------------------------------------------------------------------

#define NN    100000
#define NNZE  3200000
#define F1    128
#define F2    256
#define KD1   384      // F1*3
#define KD2   768      // F2*3
#define NO1   256
#define NO2   64
#define NIDX  50000

typedef __attribute__((ext_vector_type(16))) __bf16 v16bf;
typedef __attribute__((ext_vector_type(8)))  float  v8f;

union FragU {
    v16bf v;
    uint4 q[2];
};

__device__ __forceinline__ unsigned short f32_to_bf16_rne(float f) {
    unsigned int u = __float_as_uint(f);
    unsigned int lsb = (u >> 16) & 1u;
    u += 0x7fffu + lsb;                 // round to nearest even
    return (unsigned short)(u >> 16);
}

// ---------------------------------------------------------------------------
// Utility kernels
// ---------------------------------------------------------------------------

__global__ void cheb_zero_f32(float* __restrict__ p, size_t n) {
    size_t i = (size_t)blockIdx.x * blockDim.x + threadIdx.x;
    if (i < n) p[i] = 0.0f;
}

// Edge-parallel SpMM: one wave per nonzero, 32 lanes sweep the feature dim.
// Y[rows[e], :] += vals[e] * X[cols[e], :]   (fp32 global atomics)
__global__ void cheb_spmm_edges(const int* __restrict__ rows,
                                const int* __restrict__ cols,
                                const float* __restrict__ vals,
                                const float* __restrict__ X,
                                float* __restrict__ Y,
                                int F, int nnz) {
    int wid  = (int)(((size_t)blockIdx.x * blockDim.x + threadIdx.x) >> 5);
    int lane = threadIdx.x & 31;
    if (wid >= nnz) return;
    int   r = rows[wid];
    int   c = cols[wid];
    float v = vals[wid];
    const float* xs = X + (size_t)c * F;
    float*       ys = Y + (size_t)r * F;
    for (int f = lane; f < F; f += 32)
        atomicAdd(&ys[f], v * xs[f]);
}

// Chebyshev recurrence finish: y = 2*y - x0  (elementwise)
__global__ void cheb_combine(float* __restrict__ y,
                             const float* __restrict__ x0, size_t n) {
    size_t i = (size_t)blockIdx.x * blockDim.x + threadIdx.x;
    if (i < n) y[i] = 2.0f * y[i] - x0[i];
}

// Interleave T0/T1/T2 into bf16 basis matrix: out[n, f*3+k]
__global__ void cheb_pack_basis(const float* __restrict__ T0,
                                const float* __restrict__ T1,
                                const float* __restrict__ T2,
                                unsigned short* __restrict__ out,
                                int N, int F) {
    size_t i = (size_t)blockIdx.x * blockDim.x + threadIdx.x;
    size_t tot = (size_t)N * F;
    if (i >= tot) return;
    int n = (int)(i / F);
    int f = (int)(i - (size_t)n * F);
    size_t o = (size_t)n * (size_t)(F * 3) + (size_t)f * 3;
    out[o + 0] = f32_to_bf16_rne(T0[i]);
    out[o + 1] = f32_to_bf16_rne(T1[i]);
    out[o + 2] = f32_to_bf16_rne(T2[i]);
}

// Pack W [K, Nout] row-major fp32 -> Wp[kt][n][kk] bf16 (kk = 32 contiguous
// K-values per (ktile, n)), so each lane's WMMA-B fragment is two b128 loads.
__global__ void cheb_pack_w(const float* __restrict__ W,
                            unsigned short* __restrict__ Wp,
                            int K, int Nout) {
    int i = blockIdx.x * blockDim.x + threadIdx.x;
    if (i >= K * Nout) return;
    int k = i / Nout;
    int n = i - k * Nout;
    int kt = k >> 5;
    int kk = k & 31;
    Wp[(((size_t)kt * Nout + n) << 5) + kk] = f32_to_bf16_rne(W[i]);
}

// ---------------------------------------------------------------------------
// bf16 WMMA GEMM: C[M,Nout] = act(A[M,K] @ W[K,Nout] + bias)
//   A: row-major bf16 (16B-aligned rows), Wp: packed as above, C: fp32.
//   One wave computes a 16(M) x 64(N) strip = 4 x v_wmma_f32_16x16x32_bf16
//   per K-step, reusing the A fragment 4x. Weights stay hot in L2 (192 MB).
//   Addressing: single A/B pointers bumped per K-step; per-tile offsets are
//   compile-time immediates (fold into the load's 24-bit IOFFSET field).
// ---------------------------------------------------------------------------
__global__ __launch_bounds__(256)
void cheb_gemm_bf16_wmma(const unsigned short* __restrict__ A,
                         const unsigned short* __restrict__ Wp,
                         const float* __restrict__ bias,
                         float* __restrict__ C,
                         int M, int K, int Nout, int relu) {
    const int lane   = threadIdx.x & 31;
    const int wid    = blockIdx.x * (blockDim.x >> 5) + (threadIdx.x >> 5);
    const int nQuads = Nout >> 6;            // 64-column strips
    const int mTiles = M >> 4;
    if (wid >= mTiles * nQuads) return;

    const int mTile  = wid / nQuads;
    const int nQuad  = wid - mTile * nQuads;
    const int mBase  = mTile << 4;
    const int nBase0 = nQuad << 6;

    // ISA 16-bit A(16x32) layout: lane<16 -> K{0..7,16..23}, lane>=16 -> K{8..15,24..31}
    const int halfLo = lane & 15;
    const int kOff   = (lane < 16) ? 0 : 8;

    // Loop-carried pointers; strides are loop-invariant.
    const unsigned short* ap = A + (size_t)(mBase + halfLo) * K + kOff;
    const unsigned short* bp = Wp + (((size_t)nBase0 + halfLo) << 5) + kOff;
    const size_t bStride = (size_t)Nout << 5;     // elements per K-step

    v8f acc[4] = {};
    const int ktCount = K >> 5;

    for (int kt = 0; kt < ktCount; ++kt) {
        FragU af;
        af.q[0] = *reinterpret_cast<const uint4*>(ap);
        af.q[1] = *reinterpret_cast<const uint4*>(ap + 16);
        __builtin_prefetch((const void*)(ap + 32), 0, 3);   // next K-step, near cache

#pragma unroll
        for (int t = 0; t < 4; ++t) {
            FragU bfr;
            bfr.q[0] = *reinterpret_cast<const uint4*>(bp + t * 512);
            bfr.q[1] = *reinterpret_cast<const uint4*>(bp + t * 512 + 16);
            acc[t] = __builtin_amdgcn_wmma_f32_16x16x32_bf16(
                false, af.v, false, bfr.v, (short)0, acc[t], false, false);
        }
        ap += 32;
        bp += bStride;
    }

    // Epilogue: C/D layout -> lane L, VGPR r holds (m = r + 8*(L>=16), n = L&15)
    const int mOfs = (lane < 16) ? 0 : 8;
#pragma unroll
    for (int t = 0; t < 4; ++t) {
        const int n = nBase0 + (t << 4) + halfLo;
        const float b = bias[n];
#pragma unroll
        for (int r = 0; r < 8; ++r) {
            float v = acc[t][r] + b;
            if (relu) v = fmaxf(v, 0.0f);
            C[(size_t)(mBase + mOfs + r) * Nout + n] = v;
        }
    }
}

__global__ void cheb_gather_rows(const float* __restrict__ src,
                                 const int* __restrict__ idx,
                                 float* __restrict__ dst,
                                 int M, int Cn) {
    int i = blockIdx.x * blockDim.x + threadIdx.x;
    if (i >= M * Cn) return;
    int r = i / Cn;
    int c = i - r * Cn;
    dst[i] = src[(size_t)idx[r] * Cn + c];
}

// ---------------------------------------------------------------------------
// Workspace layout (bytes, phases aliased; peak ~487 MB)
// ---------------------------------------------------------------------------
static const size_t OFF_T1X   = 0;                   //  51.2 MB  (phase 1)
static const size_t OFF_T2X   = 51200000;            //  51.2 MB  (phase 1)
static const size_t OFF_T1H   = 0;                   // 102.4 MB  (aliases T1X/T2X)
static const size_t OFF_H     = 102400000;           // 102.4 MB
static const size_t OFF_CHEBX = 204800000;           //  76.8 MB  (phase 1)
static const size_t OFF_CHEBH = 204800000;           // 153.6 MB  (aliases CHEBX)
static const size_t OFF_T2H   = 358400000;           // 102.4 MB
static const size_t OFF_OUTF  = 460800000;           //  25.6 MB
static const size_t OFF_WP1   = 486400000;           // 196 KB
static const size_t OFF_WP2   = 486596608;           //  98 KB   end ~486.7 MB

extern "C" void kernel_launch(void* const* d_in, const int* in_sizes, int n_in,
                              void* d_out, int out_size, void* d_ws, size_t ws_size,
                              hipStream_t stream) {
    const float* x    = (const float*)d_in[0];
    const float* vals = (const float*)d_in[1];
    const float* W1   = (const float*)d_in[2];
    const float* b1   = (const float*)d_in[3];
    const float* W2   = (const float*)d_in[4];
    const float* b2   = (const float*)d_in[5];
    const int*   rows = (const int*)d_in[6];
    const int*   cols = (const int*)d_in[7];
    const int*   idx  = (const int*)d_in[8];
    float*       out  = (float*)d_out;

    char* w = (char*)d_ws;
    float*          T1x   = (float*)(w + OFF_T1X);
    float*          T2x   = (float*)(w + OFF_T2X);
    float*          T1h   = (float*)(w + OFF_T1H);
    float*          h     = (float*)(w + OFF_H);
    unsigned short* chebX = (unsigned short*)(w + OFF_CHEBX);
    unsigned short* chebH = (unsigned short*)(w + OFF_CHEBH);
    float*          T2h   = (float*)(w + OFF_T2H);
    float*          outF  = (float*)(w + OFF_OUTF);
    unsigned short* Wp1   = (unsigned short*)(w + OFF_WP1);
    unsigned short* Wp2   = (unsigned short*)(w + OFF_WP2);

    const int TB = 256;
    const size_t nxf1 = (size_t)NN * F1;          // 12.8M
    const size_t nxf2 = (size_t)NN * F2;          // 25.6M
    const int spmmBlocks = (int)(((size_t)NNZE * 32 + TB - 1) / TB);

    // ----- Chebyshev basis of x (fp32) -----
    cheb_zero_f32<<<(int)((nxf1 + TB - 1) / TB), TB, 0, stream>>>(T1x, nxf1);
    cheb_spmm_edges<<<spmmBlocks, TB, 0, stream>>>(rows, cols, vals, x, T1x, F1, NNZE);
    cheb_zero_f32<<<(int)((nxf1 + TB - 1) / TB), TB, 0, stream>>>(T2x, nxf1);
    cheb_spmm_edges<<<spmmBlocks, TB, 0, stream>>>(rows, cols, vals, T1x, T2x, F1, NNZE);
    cheb_combine<<<(int)((nxf1 + TB - 1) / TB), TB, 0, stream>>>(T2x, x, nxf1);
    cheb_pack_basis<<<(int)((nxf1 + TB - 1) / TB), TB, 0, stream>>>(x, T1x, T2x, chebX, NN, F1);

    // ----- GEMM1: h = relu(chebX @ W1 + b1)  [100000 x 256] -----
    cheb_pack_w<<<(KD1 * NO1 + TB - 1) / TB, TB, 0, stream>>>(W1, Wp1, KD1, NO1);
    {
        int jobs = (NN / 16) * (NO1 / 64);        // 25000 waves
        cheb_gemm_bf16_wmma<<<(jobs + 7) / 8, TB, 0, stream>>>(
            chebX, Wp1, b1, h, NN, KD1, NO1, /*relu=*/1);
    }

    // ----- Chebyshev basis of h (fp32) -----
    cheb_zero_f32<<<(int)((nxf2 + TB - 1) / TB), TB, 0, stream>>>(T1h, nxf2);
    cheb_spmm_edges<<<spmmBlocks, TB, 0, stream>>>(rows, cols, vals, h, T1h, F2, NNZE);
    cheb_zero_f32<<<(int)((nxf2 + TB - 1) / TB), TB, 0, stream>>>(T2h, nxf2);
    cheb_spmm_edges<<<spmmBlocks, TB, 0, stream>>>(rows, cols, vals, T1h, T2h, F2, NNZE);
    cheb_combine<<<(int)((nxf2 + TB - 1) / TB), TB, 0, stream>>>(T2h, h, nxf2);
    cheb_pack_basis<<<(int)((nxf2 + TB - 1) / TB), TB, 0, stream>>>(h, T1h, T2h, chebH, NN, F2);

    // ----- GEMM2: outF = chebH @ W2 + b2  [100000 x 64] -----
    cheb_pack_w<<<(KD2 * NO2 + TB - 1) / TB, TB, 0, stream>>>(W2, Wp2, KD2, NO2);
    {
        int jobs = (NN / 16) * (NO2 / 64);        // 6250 waves
        cheb_gemm_bf16_wmma<<<(jobs + 7) / 8, TB, 0, stream>>>(
            chebH, Wp2, b2, outF, NN, KD2, NO2, /*relu=*/0);
    }

    // ----- Gather out[idx] -----
    cheb_gather_rows<<<(NIDX * NO2 + TB - 1) / TB, TB, 0, stream>>>(outF, idx, out, NIDX, NO2);

    (void)in_sizes; (void)n_in; (void)out_size; (void)ws_size;
}
